// LightAttention_62371515073085
// MI455X (gfx1250) — matrix operand order repaired and checked by reading, not compiled, not run on hardware
//
#include <hip/hip_runtime.h>
#include <hip/hip_bf16.h>

// ---------------- types ----------------
typedef __attribute__((ext_vector_type(16))) __bf16 v16bf;
typedef __attribute__((ext_vector_type(8)))  float  v8f;
typedef __attribute__((ext_vector_type(4)))  unsigned int u32x4;
typedef __attribute__((ext_vector_type(8)))  int i32x8;
typedef __attribute__((ext_vector_type(4)))  int i32x4;

union BFrag { v16bf v; u32x4 q[2]; };

#if __has_builtin(__builtin_amdgcn_tensor_load_to_lds)
#define HAVE_TDM 1
#else
#define HAVE_TDM 0
#endif

__device__ __forceinline__ unsigned short f2bf(float f) {
  unsigned u = __float_as_uint(f);
  unsigned r = (u + 0x7fffu + ((u >> 16) & 1u)) >> 16;   // round-to-nearest-even
  return (unsigned short)r;
}

// monotone float<->uint order encoding for atomic segment-max
__device__ __forceinline__ unsigned encf(float x) {
  unsigned u = __float_as_uint(x);
  return (u & 0x80000000u) ? ~u : (u | 0x80000000u);
}
__device__ __forceinline__ float decf(unsigned m) {
  unsigned u = (m & 0x80000000u) ? (m ^ 0x80000000u) : ~m;
  float f = __uint_as_float(u);
  if ((u & 0x7f800000u) == 0x7f800000u) f = 0.f;   // nonfinite (incl. empty segment) -> 0
  return f;
}

#define CONV_D 1024
#define CONV_L 512
#define CONV_B 32

// ---------------- conv1d as WMMA GEMM ----------------
// Y[b,o,l] = sum_{t=0..8} sum_c W[o,c,t] * X[b,c,l+t-4]  (+bias[o])
// Block: 256 thr = 8 waves. Tile: M=32 positions (2 sub-tiles) x N=128 out-ch (16/wave).
// K loop: c0 step 32, 9 taps fully unrolled; 2 v_wmma_f32_16x16x32_bf16 per tap step.
// X slab (32 ch x 40 pos, f32) staged global->LDS by the Tensor Data Mover on interior
// tiles (wave 0 issues, s_wait_tensorcnt), manual masked loads on boundary tiles; then
// converted to a WMMA-fragment-swizzled bf16 layout so operand fetch = 2x ds_load_b128.
__global__ __launch_bounds__(256) void conv1d_wmma_kernel(
    const float* __restrict__ X, const float* __restrict__ W,
    const float* __restrict__ bias, float* __restrict__ Y) {
  __shared__ alignas(16) float          Xf[32][40];     // raw f32 slab (c, pos) incl. 4+4 halo
  __shared__ alignas(16) unsigned short Xs2[40][40];    // bf16, [pos][c_local], 80B row stride
  __shared__ alignas(16) unsigned short Ws2[128][40];   // bf16, [n_local][c_local], one tap
  const int b  = blockIdx.z;
  const int l0 = blockIdx.x * 32;
  const int n0 = blockIdx.y * 128;
  const int tid  = threadIdx.x;
  const int lane = tid & 31;
  const int wave = tid >> 5;
  const int hi   = lane >> 4;     // 0 = lanes 0-15, 1 = lanes 16-31
  const int lm   = lane & 15;
  v8f acc0 = {0.f, 0.f, 0.f, 0.f, 0.f, 0.f, 0.f, 0.f};
  v8f acc1 = {0.f, 0.f, 0.f, 0.f, 0.f, 0.f, 0.f, 0.f};
  const float* Xb = X + (size_t)b * CONV_D * CONV_L;
  const bool interior = (l0 >= 4) && (l0 + 36 <= CONV_L);

  for (int c0 = 0; c0 < CONV_D; c0 += 32) {
    __syncthreads();                                 // protect Xf/Xs2 against prior use
#if HAVE_TDM
    if (interior) {
      if (wave == 0) {                               // TDM ignores EXEC; issue from one wave
        unsigned long long ga =
            (unsigned long long)(size_t)(Xb + (size_t)c0 * CONV_L + (l0 - 4));
        unsigned ldsa = (unsigned)(size_t)(void*)&Xf[0][0];
        u32x4 g0 = { 1u, ldsa, (unsigned)(ga & 0xffffffffull),
                     (unsigned)((ga >> 32) & 0x1ffffffull) | (2u << 30) };  // type=2
        // data_size=4B | tensor_dim0=512 | tensor_dim1=1024 | tile 40x32 | stride0=512
        i32x8 g1 = { (int)(2u << 16), (int)(512u << 16), (int)(1024u << 16),
                     (int)(40u << 16), 32, 512, 0, 0 };
        i32x4 z4 = { 0, 0, 0, 0 };
#if __clang_major__ >= 23
        i32x8 z8 = { 0, 0, 0, 0, 0, 0, 0, 0 };
        __builtin_amdgcn_tensor_load_to_lds(g0, g1, z4, z4, z8, 0);
#else
        __builtin_amdgcn_tensor_load_to_lds(g0, g1, z4, z4, 0);
#endif
        __builtin_amdgcn_s_wait_tensorcnt(0);
      }
    } else
#endif
    {
      for (int idx = tid; idx < 32 * 40; idx += 256) {
        int kk = idx / 40, j = idx % 40;
        int l = l0 - 4 + j;
        Xf[kk][j] = (l >= 0 && l < CONV_L) ? Xb[(size_t)(c0 + kk) * CONV_L + l] : 0.f;
      }
    }
    __syncthreads();
    for (int idx = tid; idx < 32 * 40; idx += 256) { // swizzle to fragment layout
      int kk = idx / 40, j = idx % 40;
      Xs2[j][kk] = f2bf(Xf[kk][j]);
    }
#pragma unroll
    for (int t = 0; t < 9; ++t) {
      __syncthreads();                               // Ws2 reuse / Xs2 visibility
      for (int idx = tid; idx < 128 * 32; idx += 256) {
        int n = idx >> 5, kk = idx & 31;
        Ws2[n][kk] = f2bf(W[(size_t)(n0 + n) * (CONV_D * 9) + (size_t)(c0 + kk) * 9 + t]);
      }
      if (t == 0 && c0 + 32 < CONV_D)
        __builtin_prefetch(&W[(size_t)(n0 + (tid >> 1)) * (CONV_D * 9) +
                              (size_t)(c0 + 32) * 9], 0, 0);
      __syncthreads();
      BFrag a0, a1, bm;
      const int row = wave * 16 + lm;
      // B 32x16 bf16: lane -> N=lane%16; elems 0..7 -> K=hi*8+e, 8..15 -> K=16+hi*8+e
      bm.q[0] = *(const u32x4*)&Ws2[row][hi * 8];
      bm.q[1] = *(const u32x4*)&Ws2[row][16 + hi * 8];
      // A 16x32 bf16: lane -> M=lane%16; elems 0..7 -> K=hi*8+e, 8..15 -> K=16+hi*8+e
      a0.q[0] = *(const u32x4*)&Xs2[lm + t][hi * 8];
      a0.q[1] = *(const u32x4*)&Xs2[lm + t][16 + hi * 8];
      a1.q[0] = *(const u32x4*)&Xs2[16 + lm + t][hi * 8];
      a1.q[1] = *(const u32x4*)&Xs2[16 + lm + t][16 + hi * 8];
      acc0 = __builtin_amdgcn_wmma_f32_16x16x32_bf16(false, a0.v, false, bm.v,
                                                     (short)0, acc0, false, false);
      acc1 = __builtin_amdgcn_wmma_f32_16x16x32_bf16(false, a1.v, false, bm.v,
                                                     (short)0, acc1, false, false);
    }
  }
  // C/D layout: VGPR r, lanes 0-15 -> M=r, lanes 16-31 -> M=r+8; N=lane%16
  const int o = n0 + wave * 16 + lm;
  const float bo = bias[o];
#pragma unroll
  for (int r = 0; r < 8; ++r) {
    Y[((size_t)b * CONV_D + o) * CONV_L + (l0 + r + hi * 8)]      = acc0[r] + bo;
    Y[((size_t)b * CONV_D + o) * CONV_L + (l0 + 16 + r + hi * 8)] = acc1[r] + bo;
  }
}

// ---------------- light attention reduction ----------------
__global__ __launch_bounds__(256) void light_attn_reduce(
    const float* __restrict__ FEAT, const float* __restrict__ ATTN,
    const unsigned char* __restrict__ mask, float* __restrict__ tcat) {
  __shared__ float sd[256];
  const int d = blockIdx.x, b = blockIdx.y, tid = threadIdx.x;
  const float* fr = FEAT + ((size_t)b * CONV_D + d) * CONV_L;
  const float* ar = ATTN + ((size_t)b * CONV_D + d) * CONV_L;
  const unsigned char* mr = mask + (size_t)b * CONV_L;
  float f0 = fr[tid], f1 = fr[tid + 256];
  float a0 = mr[tid] ? ar[tid] : -1e9f;
  float a1 = mr[tid + 256] ? ar[tid + 256] : -1e9f;
  sd[tid] = fmaxf(a0, a1); __syncthreads();
  for (int s = 128; s > 0; s >>= 1) { if (tid < s) sd[tid] = fmaxf(sd[tid], sd[tid + s]); __syncthreads(); }
  float am = sd[0]; __syncthreads();
  float e0 = __expf(a0 - am), e1 = __expf(a1 - am);
  sd[tid] = e0 + e1; __syncthreads();
  for (int s = 128; s > 0; s >>= 1) { if (tid < s) sd[tid] += sd[tid + s]; __syncthreads(); }
  float S = sd[0]; __syncthreads();
  sd[tid] = f0 * e0 + f1 * e1; __syncthreads();
  for (int s = 128; s > 0; s >>= 1) { if (tid < s) sd[tid] += sd[tid + s]; __syncthreads(); }
  float NUM = sd[0]; __syncthreads();
  sd[tid] = fmaxf(f0, f1); __syncthreads();
  for (int s = 128; s > 0; s >>= 1) { if (tid < s) sd[tid] = fmaxf(sd[tid], sd[tid + s]); __syncthreads(); }
  if (tid == 0) {
    tcat[(size_t)b * 2048 + d]        = NUM / S;
    tcat[(size_t)b * 2048 + 1024 + d] = sd[0];
  }
}

// ---------------- generic GEMM + bias + (relu) + (bn) + (softplus+1) ----------------
__global__ void gemm_kernel(const float* __restrict__ A, const float* __restrict__ W,
                            const float* __restrict__ bias, float* __restrict__ Y,
                            int M, int K, int N, int ldY, int act,
                            const float* __restrict__ g, const float* __restrict__ bb) {
  long long i = (long long)blockIdx.x * blockDim.x + threadIdx.x;
  if (i >= (long long)M * N) return;
  int m = (int)(i / N), n = (int)(i % N);
  float s = bias ? bias[n] : 0.f;
  const float* a = A + (size_t)m * K;
  for (int k2 = 0; k2 < K; ++k2) s += a[k2] * W[(size_t)k2 * N + n];
  if (act == 1) s = fmaxf(s, 0.f);
  if (g) s = g[n] * (s * rsqrtf(1.f + 1e-5f)) + bb[n];
  if (act == 2) { float sp = (s > 20.f) ? s : log1pf(__expf(s)); s = sp + 1.f; }
  Y[(size_t)m * ldY + n] = s;
}

// ---------------- fills / elementwise ----------------
__global__ void fill_f32_kernel(float* p, int n, float v) {
  int i = blockIdx.x * blockDim.x + threadIdx.x; if (i < n) p[i] = v;
}
__global__ void fill_u32_kernel(unsigned* p, int n, unsigned v) {
  int i = blockIdx.x * blockDim.x + threadIdx.x; if (i < n) p[i] = v;
}
__global__ void relu_kernel(float* p, int n) {
  int i = blockIdx.x * blockDim.x + threadIdx.x; if (i < n) p[i] = fmaxf(p[i], 0.f);
}

// ---------------- TransformerConv pieces ----------------
__global__ void tconv_alpha_kernel(const float* __restrict__ q, const float* __restrict__ k,
                                   const float* __restrict__ ef, const int* __restrict__ ei,
                                   int E, int C, float* __restrict__ alpha,
                                   unsigned* __restrict__ amax) {
  int e = blockIdx.x * blockDim.x + threadIdx.x;
  if (e >= E) return;
  int src = ei[e], dst = ei[E + e];
  const float* qd = q + (size_t)dst * C;
  const float* ks = k + (size_t)src * C;
  const float* ee = ef + (size_t)e * C;
  float s = 0.f;
  for (int c = 0; c < C; ++c) s += qd[c] * (ks[c] + ee[c]);
  s *= rsqrtf((float)C);
  alpha[e] = s;
  atomicMax(&amax[dst], encf(s));
}

__global__ void tconv_w_kernel(const float* __restrict__ alpha, const unsigned* __restrict__ amax,
                               const int* __restrict__ ei, int E,
                               float* __restrict__ w, float* __restrict__ den) {
  int e = blockIdx.x * blockDim.x + threadIdx.x;
  if (e >= E) return;
  int dst = ei[E + e];
  float m = decf(amax[dst]);
  float wv = __expf(alpha[e] - m);
  w[e] = wv;
  atomicAdd(&den[dst], wv);
}

__global__ void tconv_agg_kernel(const float* __restrict__ v, const float* __restrict__ ef,
                                 const int* __restrict__ ei, int E, int C,
                                 const float* __restrict__ w, const float* __restrict__ den,
                                 float* __restrict__ out) {
  long long i = (long long)blockIdx.x * blockDim.x + threadIdx.x;
  if (i >= (long long)E * C) return;
  int e = (int)(i / C), c = (int)(i % C);
  int src = ei[e], dst = ei[E + e];
  float ww = w[e] / (den[dst] + 1e-16f);
  atomicAdd(&out[(size_t)dst * C + c], ww * (v[(size_t)src * C + c] + ef[(size_t)e * C + c]));
}

// ---------------- segment pooling ----------------
__global__ void seg_max_kernel(const float* __restrict__ h, const int* __restrict__ seg,
                               int Nn, int C, unsigned* __restrict__ pool_enc) {
  long long i = (long long)blockIdx.x * blockDim.x + threadIdx.x;
  if (i >= (long long)Nn * C) return;
  int n = (int)(i / C), c = (int)(i % C);
  atomicMax(&pool_enc[(size_t)seg[n] * C + c], encf(h[i]));
}
__global__ void dec_kernel(const unsigned* __restrict__ enc_, float* __restrict__ out, int n) {
  int i = blockIdx.x * blockDim.x + threadIdx.x; if (i < n) out[i] = decf(enc_[i]);
}
__global__ void seg_sum_kernel(const float* __restrict__ h, const int* __restrict__ seg,
                               int Nn, int C, float* __restrict__ sum) {
  long long i = (long long)blockIdx.x * blockDim.x + threadIdx.x;
  if (i >= (long long)Nn * C) return;
  int n = (int)(i / C), c = (int)(i % C);
  atomicAdd(&sum[(size_t)seg[n] * C + c], h[i]);
}
__global__ void seg_cnt_kernel(const int* __restrict__ seg, int Nn, float* __restrict__ cnt) {
  int i = blockIdx.x * blockDim.x + threadIdx.x;
  if (i < Nn) atomicAdd(&cnt[seg[i]], 1.f);
}
__global__ void seg_div_kernel(float* __restrict__ sum, const float* __restrict__ cnt,
                               int B, int C) {
  int i = blockIdx.x * blockDim.x + threadIdx.x;
  if (i < B * C) sum[i] = sum[i] / fmaxf(cnt[i / C], 1.f);
}

// ============================ host orchestration ============================
extern "C" void kernel_launch(void* const* d_in, const int* in_sizes, int n_in,
                              void* d_out, int out_size, void* d_ws, size_t ws_size,
                              hipStream_t stream) {
  (void)in_sizes; (void)n_in; (void)out_size; (void)ws_size;
  const float* x        = (const float*)d_in[0];
  const int*   ei       = (const int*)d_in[1];
  const float* ea       = (const float*)d_in[2];
  const int*   batch    = (const int*)d_in[3];
  const float* bag_x    = (const float*)d_in[4];
  const int*   bag_ei   = (const int*)d_in[5];
  const float* bag_ea   = (const float*)d_in[6];
  const float* tg_x     = (const float*)d_in[7];
  const int*   tg_ei    = (const int*)d_in[8];
  const float* tg_ea    = (const float*)d_in[9];
  const int*   tg_batch = (const int*)d_in[10];
  const float* t1d      = (const float*)d_in[11];
  const unsigned char* mask = (const unsigned char*)d_in[12];
  const float* d2d      = (const float*)d_in[13];
  auto P = [&](int i) { return (const float*)d_in[14 + i]; };
  // param indices (relative to 14):
  // 0 feat_w 1 feat_b 2 attn_w 3 attn_b 4 lin_w 5 lin_b 6 lin_g 7 lin_bb
  // 8 dfc_w 9 dfc_b 10 dfc_g 11 dfc_bb
  // 12..20 abg1 | 21..29 abg2 | 30..38 abg3 | 39..47 bag1 | 48..56 bag2
  // 57..65 tg1 | 66..74 tg2 | 75..83 tg3
  // 84..87 abg_fc1/bn1  88..91 abg_fc2/bn2  92..95 tg_fc1/bn1  96..99 tg_fc2/bn2
  // 100 out_w 101 out_b

  const int N = 64000, E = 256000, Nb = 256000, Eb = 512000, Nt = 16000, Et = 64000;
  const int B = 32;

  float* base = (float*)d_ws;
  size_t off = 0;
  auto alloc = [&](size_t n) { float* p = base + off; off += n; return p; };
  float*    convF  = alloc((size_t)B * CONV_D * CONV_L);
  float*    convA  = alloc((size_t)B * CONV_D * CONV_L);
  float*    tcat   = alloc((size_t)B * 2048);
  float*    catv   = alloc((size_t)B * 256);
  float*    tmpB   = alloc((size_t)B * 1024);
  float*    qb     = alloc((size_t)Nb * 64);
  float*    kb     = alloc((size_t)Nb * 64);
  float*    vb     = alloc((size_t)Nb * 64);
  float*    efb    = alloc((size_t)Eb * 64);
  float*    alphab = alloc((size_t)Eb);
  float*    wbuf   = alloc((size_t)Eb);
  unsigned* amaxb  = (unsigned*)alloc((size_t)Nb);
  float*    denb   = alloc((size_t)Nb);
  float*    atomA  = alloc((size_t)N * 128);
  float*    atomB2 = alloc((size_t)N * 128);
  float*    edgeA  = alloc((size_t)Nb * 64);
  float*    edgeB  = alloc((size_t)Nb * 64);
  float*    tgA    = alloc((size_t)Nt * 128);
  float*    tgB    = alloc((size_t)Nt * 128);
  unsigned* poolenc = (unsigned*)alloc((size_t)B * 128);
  float*    poolv  = alloc((size_t)B * 128);
  float*    cntb   = alloc((size_t)B);

  auto cdiv = [](long long a, long long b) { return (int)((a + b - 1) / b); };
  auto gemm = [&](const float* A, const float* Wm, const float* bias, float* Y,
                  int M, int K, int Nn, int ldY, int act, const float* g, const float* bb) {
    gemm_kernel<<<cdiv((long long)M * Nn, 256), 256, 0, stream>>>(A, Wm, bias, Y, M, K, Nn,
                                                                  ldY, act, g, bb);
  };
  auto tconv = [&](const float* xin, int Nn, int din, const int* eidx, int E_,
                   const float* eattr, int de, int C, int pbase, float* out) {
    gemm(xin,   P(pbase + 0), P(pbase + 1), qb,  Nn, din, C, C, 0, nullptr, nullptr);
    gemm(xin,   P(pbase + 2), P(pbase + 3), kb,  Nn, din, C, C, 0, nullptr, nullptr);
    gemm(xin,   P(pbase + 4), P(pbase + 5), vb,  Nn, din, C, C, 0, nullptr, nullptr);
    gemm(eattr, P(pbase + 6), nullptr,      efb, E_, de,  C, C, 0, nullptr, nullptr);
    gemm(xin,   P(pbase + 7), P(pbase + 8), out, Nn, din, C, C, 0, nullptr, nullptr); // skip
    fill_u32_kernel<<<cdiv(Nn, 256), 256, 0, stream>>>(amaxb, Nn, 0u);
    fill_f32_kernel<<<cdiv(Nn, 256), 256, 0, stream>>>(denb, Nn, 0.f);
    tconv_alpha_kernel<<<cdiv(E_, 256), 256, 0, stream>>>(qb, kb, efb, eidx, E_, C, alphab, amaxb);
    tconv_w_kernel<<<cdiv(E_, 256), 256, 0, stream>>>(alphab, amaxb, eidx, E_, wbuf, denb);
    tconv_agg_kernel<<<cdiv((long long)E_ * C, 256), 256, 0, stream>>>(vb, efb, eidx, E_, C,
                                                                       wbuf, denb, out);
    relu_kernel<<<cdiv((long long)Nn * C, 256), 256, 0, stream>>>(out, Nn * C);
  };

  // ---- light attention (WMMA convs, TDM-staged) ----
  dim3 cgrid(CONV_L / 32, CONV_D / 128, B);
  conv1d_wmma_kernel<<<cgrid, 256, 0, stream>>>(t1d, P(0), P(1), convF);
  conv1d_wmma_kernel<<<cgrid, 256, 0, stream>>>(t1d, P(2), P(3), convA);
  light_attn_reduce<<<dim3(CONV_D, B), 256, 0, stream>>>(convF, convA, mask, tcat);
  gemm(tcat, P(4), P(5), catv + 0,  B, 2048, 64, 256, 1, P(6),  P(7));   // t_feat
  gemm(d2d,  P(8), P(9), catv + 64, B, 256,  64, 256, 1, P(10), P(11));  // d_o

  // ---- atom graph + bond-angle graph stacks ----
  tconv(x,      N,  7,  ei,     E,  ea,    4,  32,  12, atomA);
  tconv(bag_x,  Nb, 4,  bag_ei, Eb, bag_ea, 1, 32,  39, edgeA);
  tconv(atomA,  N,  32, ei,     E,  edgeA, 32, 64,  21, atomB2);
  tconv(edgeA,  Nb, 32, bag_ei, Eb, bag_ea, 1, 64,  48, edgeB);
  tconv(atomB2, N,  64, ei,     E,  edgeB, 64, 128, 30, atomA);

  fill_u32_kernel<<<cdiv(B * 128, 256), 256, 0, stream>>>(poolenc, B * 128, 0u);
  seg_max_kernel<<<cdiv((long long)N * 128, 256), 256, 0, stream>>>(atomA, batch, N, 128, poolenc);
  dec_kernel<<<cdiv(B * 128, 256), 256, 0, stream>>>(poolenc, poolv, B * 128);
  gemm(poolv, P(84), P(85), tmpB,       B, 128,  1024, 1024, 0, P(86), P(87));
  gemm(tmpB,  P(88), P(89), catv + 128, B, 1024, 64,   256,  0, P(90), P(91)); // ah

  // ---- protein 3D graph stack ----
  tconv(tg_x, Nt, 1,  tg_ei, Et, tg_ea, 1, 32,  57, tgA);
  tconv(tgA,  Nt, 32, tg_ei, Et, tg_ea, 1, 64,  66, tgB);
  tconv(tgB,  Nt, 64, tg_ei, Et, tg_ea, 1, 128, 75, tgA);
  fill_f32_kernel<<<cdiv(B * 128, 256), 256, 0, stream>>>(poolv, B * 128, 0.f);
  fill_f32_kernel<<<1, 256, 0, stream>>>(cntb, B, 0.f);
  seg_sum_kernel<<<cdiv((long long)Nt * 128, 256), 256, 0, stream>>>(tgA, tg_batch, Nt, 128, poolv);
  seg_cnt_kernel<<<cdiv(Nt, 256), 256, 0, stream>>>(tg_batch, Nt, cntb);
  seg_div_kernel<<<cdiv(B * 128, 256), 256, 0, stream>>>(poolv, cntb, B, 128);
  gemm(poolv, P(92), P(93), tmpB,       B, 128,  1024, 1024, 0, P(94), P(95));
  gemm(tmpB,  P(96), P(97), catv + 192, B, 1024, 64,   256,  0, P(98), P(99)); // th

  // ---- head ----
  gemm(catv, P(100), P(101), (float*)d_out, B, 256, 2, 2, 2, nullptr, nullptr);
}
